// ProjectedBilinear_50500225466438
// MI455X (gfx1250) — compile-verified, matrix operand order checked by reading, not run on hardware
//
#include <hip/hip_runtime.h>

typedef __attribute__((ext_vector_type(16))) _Float16 v16h;
typedef __attribute__((ext_vector_type(8)))  _Float16 v8h;
typedef __attribute__((ext_vector_type(8)))  float    v8f;
typedef __attribute__((ext_vector_type(2)))  float    v2f;

#define N_ROWS 8192
#define D_DIM  1024
#define P_DIM  64

// ---------------------------------------------------------------------------
// Kernel 1: P = X @ W^T + b   (8192x1024 @ 1024x64), full fp32 via
// V_WMMA_F32_16X16X4_F32. Each wave computes a 16x64 strip of P (4 N-tiles),
// looping K in steps of 4. Writes P in f32 (for kernel 2) and f16 (for GEMM3).
// A-frag (16x4 f32): lane<16 -> K={k0,k0+1}, lane>=16 -> K={k0+2,k0+3}, row=m0+(l&15).
// B-frag mirrored: same K pattern, row of W = n0+(l&15).
// ---------------------------------------------------------------------------
__global__ __launch_bounds__(256)
void proj_wmma_kernel(const float* __restrict__ X, const float* __restrict__ W,
                      const float* __restrict__ bias,
                      float* __restrict__ P32, _Float16* __restrict__ P16)
{
    const int lane = threadIdx.x & 31;
    const int wave = threadIdx.x >> 5;
    const int lh   = lane & 15;
    const int hi   = lane >> 4;
    const int m0   = (blockIdx.x * 8 + wave) * 16;

    v8f acc[4] = {};
    const float* xrow = X + (size_t)(m0 + lh) * D_DIM + 2 * hi;
    const float* wbase = W + (size_t)lh * D_DIM + 2 * hi;

    for (int k0 = 0; k0 < D_DIM; k0 += 4) {
        v2f a = *(const v2f*)(xrow + k0);
#pragma unroll
        for (int nt = 0; nt < 4; ++nt) {
            v2f bf = *(const v2f*)(wbase + (size_t)(nt * 16) * D_DIM + k0);
            acc[nt] = __builtin_amdgcn_wmma_f32_16x16x4_f32(
                false, a, false, bf, (short)0, acc[nt], false, false);
        }
    }

#pragma unroll
    for (int nt = 0; nt < 4; ++nt) {
        const int col = nt * 16 + lh;
        const float bv = bias[col];
#pragma unroll
        for (int v = 0; v < 8; ++v) {
            const int row = m0 + v + 8 * hi;
            const float val = acc[nt][v] + bv;
            P32[row * P_DIM + col] = val;
            P16[row * P_DIM + col] = (_Float16)val;
        }
    }
}

// ---------------------------------------------------------------------------
// Kernel 2: Q16 = f16(P32 @ C)   (8192x64 @ 64x64) — 67 MFLOP, plain VALU.
// C staged in padded LDS (65-float stride -> conflict-free column reads).
// ---------------------------------------------------------------------------
__global__ __launch_bounds__(256)
void compat_q_kernel(const float* __restrict__ P32, const float* __restrict__ C,
                     _Float16* __restrict__ Q16)
{
    __shared__ float Cs[64][65];
    for (int i = threadIdx.x; i < 64 * 64; i += 256)
        Cs[i >> 6][i & 63] = C[i];
    __syncthreads();

    const int r = blockIdx.x * 4 + (threadIdx.x >> 6);
    const int j = threadIdx.x & 63;
    const float* prow = P32 + (size_t)r * P_DIM;
    float acc = 0.f;
#pragma unroll
    for (int k = 0; k < 64; ++k)
        acc += prow[k] * Cs[k][j];
    Q16[(size_t)r * P_DIM + j] = (_Float16)acc;
}

// ---------------------------------------------------------------------------
// Kernel 3: out = Q16 @ P16^T + cb  (8192x64 @ 64x8192 -> 8192x8192 f32).
// Bandwidth-bound on the 256 MB output. Wave tile 64x64 (4x4 of 16x16),
// block = 4 waves (2x2) -> 128x128 per block, grid 64x64.
// K=64 -> two v_wmma_f32_16x16x32_f16 steps per accumulator (32 WMMA/wave).
// A and B fragments: two contiguous 16B chunks per lane per fragment.
// ---------------------------------------------------------------------------
__global__ __launch_bounds__(128)
void bilinear_out_kernel(const _Float16* __restrict__ Q16,
                         const _Float16* __restrict__ P16,
                         const float* __restrict__ cbp,
                         float* __restrict__ out)
{
    const int lane = threadIdx.x & 31;
    const int wave = threadIdx.x >> 5;
    const int lh   = lane & 15;
    const int hi   = lane >> 4;
    const int m_base = blockIdx.y * 128 + (wave >> 1) * 64;
    const int n_base = blockIdx.x * 128 + (wave & 1) * 64;
    const float cb = cbp[0];

    v8f acc[4][4] = {};

#pragma unroll
    for (int ks = 0; ks < 2; ++ks) {
        const int ko = ks * 32 + hi * 8;   // half-element offset within the row
        v16h a[4], b[4];
#pragma unroll
        for (int mi = 0; mi < 4; ++mi) {
            const _Float16* p = Q16 + (size_t)(m_base + mi * 16 + lh) * P_DIM + ko;
            v8h lo  = *(const v8h*)p;
            v8h hi8 = *(const v8h*)(p + 16);
            a[mi] = __builtin_shufflevector(lo, hi8, 0,1,2,3,4,5,6,7,8,9,10,11,12,13,14,15);
        }
#pragma unroll
        for (int ni = 0; ni < 4; ++ni) {
            const _Float16* p = P16 + (size_t)(n_base + ni * 16 + lh) * P_DIM + ko;
            v8h lo  = *(const v8h*)p;
            v8h hi8 = *(const v8h*)(p + 16);
            b[ni] = __builtin_shufflevector(lo, hi8, 0,1,2,3,4,5,6,7,8,9,10,11,12,13,14,15);
        }
#pragma unroll
        for (int mi = 0; mi < 4; ++mi)
#pragma unroll
            for (int ni = 0; ni < 4; ++ni)
                acc[mi][ni] = __builtin_amdgcn_wmma_f32_16x16x32_f16(
                    false, a[mi], false, b[ni], (short)0, acc[mi][ni], false, false);
    }

#pragma unroll
    for (int mi = 0; mi < 4; ++mi)
#pragma unroll
        for (int ni = 0; ni < 4; ++ni) {
            const int col = n_base + ni * 16 + lh;
#pragma unroll
            for (int v = 0; v < 8; ++v) {
                const int row = m_base + mi * 16 + v + 8 * hi;
                __builtin_nontemporal_store(acc[mi][ni][v] + cb,
                                            out + (size_t)row * N_ROWS + col);
            }
        }
}

extern "C" void kernel_launch(void* const* d_in, const int* in_sizes, int n_in,
                              void* d_out, int out_size, void* d_ws, size_t ws_size,
                              hipStream_t stream) {
    const float* X  = (const float*)d_in[0];   // inputs   (8192,1024)
    const float* W  = (const float*)d_in[1];   // proj_w   (64,1024)
    const float* b  = (const float*)d_in[2];   // proj_b   (64,)
    const float* C  = (const float*)d_in[3];   // compat_w (64,64)
    const float* cb = (const float*)d_in[4];   // compat_b scalar
    float* out = (float*)d_out;

    char* ws = (char*)d_ws;
    float*    P32 = (float*)ws;                                        // 2 MB
    _Float16* P16 = (_Float16*)(ws + (size_t)N_ROWS * P_DIM * 4);      // 1 MB
    _Float16* Q16 = (_Float16*)(ws + (size_t)N_ROWS * P_DIM * 6);      // 1 MB

    proj_wmma_kernel<<<N_ROWS / 128, 256, 0, stream>>>(X, W, b, P32, P16);
    compat_q_kernel<<<N_ROWS / 4, 256, 0, stream>>>(P32, C, Q16);
    dim3 grid(N_ROWS / 128, N_ROWS / 128);
    bilinear_out_kernel<<<grid, 128, 0, stream>>>(Q16, P16, cb, out);
}
